// SRULayer_37864431681610
// MI455X (gfx1250) — matrix operand order repaired and checked by reading, not compiled, not run on hardware
//
#include <hip/hip_runtime.h>

typedef __attribute__((ext_vector_type(16))) __bf16 v16bf;
typedef __attribute__((ext_vector_type(8)))  float  v8f;
typedef __attribute__((ext_vector_type(4)))  __bf16 v4bf;

#define D_DIM 1024
#define B_DIM 8
#define L_DIM 2048
#define M_DIM (L_DIM * B_DIM)   // 16384 GEMM rows
#define N_DIM (3 * D_DIM)       // 3072
#define K_DIM D_DIM             // 1024
#define TK 32                   // K per WMMA step (bf16)
#define LDK 40                  // padded LDS K-stride (elements) -> 80B rows, conflict-free b128
#define BM 128                  // block tile M
#define BN 256                  // block tile N
#define NKT (K_DIM / TK)        // 32 K-tiles

// ---------------------------------------------------------------------------
// LayerNorm: one block per (l,b) row. Emits f32 (for output mix) + bf16 (GEMM A).
// ---------------------------------------------------------------------------
__global__ __launch_bounds__(256)
void ln_kernel(const float* __restrict__ x, const float* __restrict__ g,
               const float* __restrict__ bta, float* __restrict__ xn,
               __bf16* __restrict__ xnb) {
    const int row = blockIdx.x;
    const int t   = threadIdx.x;
    const float4 v = ((const float4*)(x + (size_t)row * D_DIM))[t];

    __shared__ float rs[256], rs2[256];
    rs[t]  = v.x + v.y + v.z + v.w;
    rs2[t] = v.x * v.x + v.y * v.y + v.z * v.z + v.w * v.w;
    __syncthreads();
    for (int off = 128; off > 0; off >>= 1) {
        if (t < off) { rs[t] += rs[t + off]; rs2[t] += rs2[t + off]; }
        __syncthreads();
    }
    const float mu   = rs[0] * (1.0f / D_DIM);
    const float var  = rs2[0] * (1.0f / D_DIM) - mu * mu;
    const float rstd = rsqrtf(var + 1e-5f);

    const float4 gg = ((const float4*)g)[t];
    const float4 bb = ((const float4*)bta)[t];
    float4 o;
    o.x = (v.x - mu) * rstd * gg.x + bb.x;
    o.y = (v.y - mu) * rstd * gg.y + bb.y;
    o.z = (v.z - mu) * rstd * gg.z + bb.z;
    o.w = (v.w - mu) * rstd * gg.w + bb.w;
    ((float4*)(xn + (size_t)row * D_DIM))[t] = o;

    v4bf ob;
    ob[0] = (__bf16)o.x; ob[1] = (__bf16)o.y; ob[2] = (__bf16)o.z; ob[3] = (__bf16)o.w;
    *(v4bf*)(xnb + (size_t)row * D_DIM + t * 4) = ob;
}

// ---------------------------------------------------------------------------
// W f32 -> bf16 (elementwise, x4 vectorized)
// ---------------------------------------------------------------------------
__global__ __launch_bounds__(256)
void wconv_kernel(const float* __restrict__ w, __bf16* __restrict__ wb) {
    const size_t i = (size_t)blockIdx.x * 256 + threadIdx.x;
    const float4 v = ((const float4*)w)[i];
    v4bf o;
    o[0] = (__bf16)v.x; o[1] = (__bf16)v.y; o[2] = (__bf16)v.z; o[3] = (__bf16)v.w;
    ((v4bf*)wb)[i] = o;
}

// ---------------------------------------------------------------------------
// Async global->LDS issue for one K-tile (per-thread share: A 2 chunks, B 4).
// Uses gfx1250 GLOBAL_LOAD_ASYNC_TO_LDS_B128 (ASYNCcnt-tracked DMA into LDS).
// ---------------------------------------------------------------------------
__device__ __forceinline__
void issue_tile_async(const __bf16* __restrict__ A, const __bf16* __restrict__ Bw,
                      int m0, int n0, int t, int kt,
                      unsigned ldsA, unsigned ldsB) {
    const size_t kof = (size_t)kt * TK;
#pragma unroll
    for (int i = 0; i < 2; ++i) {                       // A tile: 128 x 32
        const int c = t + i * 256, row = c >> 2, col = (c & 3) * 8;
        const unsigned long long g =
            (unsigned long long)(uintptr_t)(A + (size_t)(m0 + row) * K_DIM + kof + col);
        const unsigned l = ldsA + (unsigned)((row * LDK + col) * 2);
        asm volatile("global_load_async_to_lds_b128 %0, %1, off"
                     :: "v"(l), "v"(g) : "memory");
    }
#pragma unroll
    for (int i = 0; i < 4; ++i) {                       // B tile: 256 x 32
        const int c = t + i * 256, row = c >> 2, col = (c & 3) * 8;
        const unsigned long long g =
            (unsigned long long)(uintptr_t)(Bw + (size_t)(n0 + row) * K_DIM + kof + col);
        const unsigned l = ldsB + (unsigned)((row * LDK + col) * 2);
        asm volatile("global_load_async_to_lds_b128 %0, %1, off"
                     :: "v"(l), "v"(g) : "memory");
    }
}

// ---------------------------------------------------------------------------
// bf16 WMMA GEMM: ufr[M,N] = xnorm_bf16[M,K] * W_bf16[N,K]^T  (f32 accumulate)
// Block tile 128x256, 8 waves as 2(M) x 4(N), wave tile 64x64 (4x4 of 16x16).
// Double-buffered LDS filled by async-to-LDS DMA; 16 WMMAs per 16 ds_loads.
// ---------------------------------------------------------------------------
__global__ __launch_bounds__(256)
void gemm_kernel(const __bf16* __restrict__ A, const __bf16* __restrict__ Bw,
                 float* __restrict__ C) {
    __shared__ __bf16 As[2][BM * LDK];
    __shared__ __bf16 Bs[2][BN * LDK];

    const int t    = threadIdx.x;
    const int n0   = blockIdx.x * BN;
    const int m0   = blockIdx.y * BM;
    const int wave = t >> 5;
    const int lane = t & 31;
    const int wm   = wave >> 2;          // 0..1 : 64-row wave tile in M
    const int wn   = wave & 3;           // 0..3 : 64-col wave tile in N
    const int lrow = lane & 15;
    const int khalf = lane >> 4;         // 0/1

    const unsigned ldsA0 = (unsigned)(size_t)&As[0][0];
    const unsigned ldsA1 = (unsigned)(size_t)&As[1][0];
    const unsigned ldsB0 = (unsigned)(size_t)&Bs[0][0];
    const unsigned ldsB1 = (unsigned)(size_t)&Bs[1][0];

    v8f acc[4][4] = {};

    // Prefetch K-tile 0 into buffer 0 (6 async b128 per thread -> ASYNCcnt=6)
    issue_tile_async(A, Bw, m0, n0, t, 0, ldsA0, ldsB0);

    for (int kt = 0; kt < NKT; ++kt) {
        const int cur = kt & 1;
        if (kt + 1 < NKT) {
            // Next tile's DMA in flight during this tile's WMMAs.
            issue_tile_async(A, Bw, m0, n0, t, kt + 1,
                             cur ? ldsA0 : ldsA1, cur ? ldsB0 : ldsB1);
            // Async loads complete in order: <=6 pending => tile kt has landed.
            asm volatile("s_wait_asynccnt 0x6" ::: "memory");
        } else {
            asm volatile("s_wait_asynccnt 0x0" ::: "memory");
        }
        __syncthreads();   // all waves' share of tile kt is in LDS

        const __bf16* __restrict__ as = &As[cur][0];
        const __bf16* __restrict__ bs = &Bs[cur][0];

        // A fragments: lane m(+16) holds K = khalf*8.. and 16+khalf*8.. (16B each)
        v16bf afrag[4];
#pragma unroll
        for (int mt = 0; mt < 4; ++mt) {
            const int ar = wm * 64 + mt * 16 + lrow;
            union { v16bf v; uint4 u[2]; } tmp;
            tmp.u[0] = *(const uint4*)&as[ar * LDK + khalf * 8];
            tmp.u[1] = *(const uint4*)&as[ar * LDK + 16 + khalf * 8];
            afrag[mt] = tmp.v;
        }
        // B fragments: lane n(+16) holds K = khalf*8.. and 16+khalf*8.. (16B each)
        v16bf bfrag[4];
#pragma unroll
        for (int nt = 0; nt < 4; ++nt) {
            const int br = wn * 64 + nt * 16 + lrow;
            union { v16bf v; uint4 u[2]; } tmp;
            tmp.u[0] = *(const uint4*)&bs[br * LDK + khalf * 8];
            tmp.u[1] = *(const uint4*)&bs[br * LDK + 16 + khalf * 8];
            bfrag[nt] = tmp.v;
        }
#pragma unroll
        for (int mt = 0; mt < 4; ++mt)
#pragma unroll
            for (int nt = 0; nt < 4; ++nt)
                acc[mt][nt] = __builtin_amdgcn_wmma_f32_16x16x32_bf16(
                    false, afrag[mt], false, bfrag[nt],
                    (short)0, acc[mt][nt], false, false);

        __syncthreads();   // all waves done reading buf cur before DMA refills it
    }

    // Epilogue: C/D layout -> lane l, vgpr r maps to (M = 8*(l/16)+r, N = l%16)
#pragma unroll
    for (int mt = 0; mt < 4; ++mt)
#pragma unroll
        for (int nt = 0; nt < 4; ++nt) {
            const int gn = n0 + wn * 64 + nt * 16 + lrow;
#pragma unroll
            for (int r = 0; r < 8; ++r) {
                const int gm = m0 + wm * 64 + mt * 16 + khalf * 8 + r;
                C[(size_t)gm * N_DIM + gn] = acc[mt][nt][r];
            }
        }
}

// ---------------------------------------------------------------------------
// Fused gates + sequential scan + output mix + residual. One thread per (b,d).
// ---------------------------------------------------------------------------
__global__ __launch_bounds__(256)
void scan_kernel(const float* __restrict__ x, const float* __restrict__ c0,
                 const float* __restrict__ ufr, const float* __restrict__ bias,
                 const float* __restrict__ xn, float* __restrict__ out,
                 float* __restrict__ lastc) {
    const int idx = blockIdx.x * 256 + threadIdx.x;   // b*D + d
    const int d   = idx & (D_DIM - 1);
    float c = c0[idx];
    const float bu  = bias[d];
    const float bfg = bias[D_DIM + d];
    const float brg = bias[2 * D_DIM + d];
    for (int l = 0; l < L_DIM; ++l) {
        const size_t rbase = ((size_t)l * B_DIM + (idx >> 10)) * N_DIM;
        const float u = ufr[rbase + d] + bu;
        const float f = ufr[rbase + D_DIM + d] + bfg;
        const float r = ufr[rbase + 2 * D_DIM + d] + brg;
        const float fg = 1.0f / (1.0f + __expf(-f));
        const float rg = 1.0f / (1.0f + __expf(-r));
        const float ut = tanhf(u);
        c = fg * c + (1.0f - fg) * ut;
        const size_t xi = ((size_t)l * B_DIM + (idx >> 10)) * D_DIM + d;
        const float h = rg * tanhf(c) + (1.0f - rg) * xn[xi];
        out[xi] = x[xi] + h;
    }
    lastc[idx] = c;
}

// ---------------------------------------------------------------------------
extern "C" void kernel_launch(void* const* d_in, const int* in_sizes, int n_in,
                              void* d_out, int out_size, void* d_ws, size_t ws_size,
                              hipStream_t stream) {
    const float* x    = (const float*)d_in[0];
    const float* c0   = (const float*)d_in[1];
    const float* W    = (const float*)d_in[2];
    const float* bias = (const float*)d_in[3];
    const float* ln_g = (const float*)d_in[4];
    const float* ln_b = (const float*)d_in[5];

    char* ws = (char*)d_ws;
    const size_t xn_bytes  = (size_t)M_DIM * D_DIM * sizeof(float);   // 64 MB
    const size_t xnb_bytes = (size_t)M_DIM * D_DIM * sizeof(__bf16);  // 32 MB
    const size_t wb_bytes  = (size_t)N_DIM * K_DIM * sizeof(__bf16);  //  6 MB
    float*  xn  = (float*)ws;
    __bf16* xnb = (__bf16*)(ws + xn_bytes);
    __bf16* wb  = (__bf16*)(ws + xn_bytes + xnb_bytes);
    float*  ufr = (float*)(ws + xn_bytes + xnb_bytes + wb_bytes);     // 192 MB

    float* out   = (float*)d_out;
    float* lastc = out + (size_t)M_DIM * D_DIM;

    ln_kernel<<<M_DIM, 256, 0, stream>>>(x, ln_g, ln_b, xn, xnb);
    wconv_kernel<<<(N_DIM * K_DIM / 4) / 256, 256, 0, stream>>>(W, wb);
    gemm_kernel<<<dim3(N_DIM / BN, M_DIM / BM), 256, 0, stream>>>(xnb, wb, ufr);
    scan_kernel<<<(B_DIM * D_DIM) / 256, 256, 0, stream>>>(x, c0, ufr, bias, xn, out, lastc);
}
